// SequentialModel_90993177133363
// MI455X (gfx1250) — compile-verified, matrix operand order checked by reading, not run on hardware
//
#include <hip/hip_runtime.h>
#include <hip/hip_bf16.h>
#include <cmath>

// Problem constants (match reference)
#define WB  128
#define WT  2048
#define WH  128
#define WNI 64
#define WNO 64

typedef __bf16 bf16;
typedef bf16  v16bf __attribute__((ext_vector_type(16)));
typedef float v8f   __attribute__((ext_vector_type(8)));

// ---------------- WMMA helpers (wave32, 16x16x32 bf16 -> f32) ----------------

__device__ __forceinline__ v8f wmma_bf16(v16bf a, v16bf b, v8f c) {
  return __builtin_amdgcn_wmma_f32_16x16x32_bf16(false, a, false, b, (short)0, c,
                                                 false, false);
}

// 16x32 bf16 operand tile: lane L reads 16 contiguous bf16 from row (L%16),
// k-offset (L/16)*16.  Works for A (activation rows) and for B when the
// weight is stored row-major as W[n][k] (computing X @ W^T).
__device__ __forceinline__ v16bf ld_tile16(const bf16* base, int ld, int lane) {
  const bf16* p = base + (lane & 15) * ld + ((lane >> 4) << 4);
  return *reinterpret_cast<const v16bf*>(p);
}

// 16x16 f32 D tile store per documented C/D layout.
__device__ __forceinline__ void st_tile_f32(float* base, int ld, int lane, v8f d) {
  const int col = lane & 15;
  const int r0  = (lane >> 4) << 3;
#pragma unroll
  for (int i = 0; i < 8; ++i) base[(size_t)(r0 + i) * ld + col] = d[i];
}

__device__ __forceinline__ void st_tile_bf16(bf16* base, int ld, int lane, v8f d) {
  const int col = lane & 15;
  const int r0  = (lane >> 4) << 3;
#pragma unroll
  for (int i = 0; i < 8; ++i) base[(size_t)(r0 + i) * ld + col] = (bf16)d[i];
}

__device__ __forceinline__ float sigmoidf_fast(float x) {
  return 1.0f / (1.0f + __expf(-x));
}

// ---- CDNA5 async copy: global -> LDS, tracked by ASYNCcnt ----
__device__ __forceinline__ void async_copy_b64(const bf16* gsrc, bf16* ldst) {
  unsigned int       lds = (unsigned int)(uintptr_t)ldst;  // LDS byte offset
  unsigned long long ga  = (unsigned long long)(uintptr_t)gsrc;
  asm volatile("global_load_async_to_lds_b64 %0, %1, off"
               :: "v"(lds), "v"(ga) : "memory");
}

__device__ __forceinline__ void wait_async0() {
  asm volatile("s_wait_asynccnt 0" ::: "memory");
}

// ---------------- Kernel 1a: f32 -> bf16 (weights, small) ----------------

__global__ void f32_to_bf16_kernel(const float* __restrict__ src,
                                   bf16* __restrict__ dst, int n) {
  int i = blockIdx.x * blockDim.x + threadIdx.x;
  if (i < n) dst[i] = (bf16)src[i];
}

// ---------------- Kernel 1b: f32 -> bf16 (X, vectorized) ----------------

__global__ void x_to_bf16_kernel(const float* __restrict__ src,
                                 bf16* __restrict__ dst, int n4) {
  int i = blockIdx.x * blockDim.x + threadIdx.x;
  if (i < n4) {
    const float4 v = reinterpret_cast<const float4*>(src)[i];
    bf16* d = dst + (size_t)i * 4;
    d[0] = (bf16)v.x; d[1] = (bf16)v.y; d[2] = (bf16)v.z; d[3] = (bf16)v.w;
  }
}

// ---------------- Kernel 2: fused GRU scan ----------------
// Grid: B/16 = 8 workgroups; block = 32x8 (8 wave32 waves).
// Weights resident in LDS; per step:
//   phase1: hp tile (1/wave, serial chain) + xg tiles (3/wave, off-chain)
//   (issue async X copy for t+1, double-buffered)
//   phase2: hg tiles (3/wave)
//   phase3: elementwise gates, h update, hidden store

// LDS layout (element offsets)
constexpr int OFF_WREC = 0;                      // 128*128 bf16
constexpr int OFF_WHH  = OFF_WREC + WH * WH;     // 384*128 bf16
constexpr int OFF_WIH  = OFF_WHH + 3 * WH * WH;  // 384*64  bf16
constexpr int OFF_H    = OFF_WIH + 3 * WH * WNI; // 16*128  bf16
constexpr int OFF_HPB  = OFF_H + 16 * WH;        // 16*128  bf16
constexpr int OFF_XT   = OFF_HPB + 16 * WH;      // 2 x (16*64) bf16 double buffer
constexpr int NBF      = OFF_XT + 2 * 16 * WNI;  // = 96256 bf16 (192512 B)

constexpr int FOFF_BIH = 0;                      // 384 f32
constexpr int FOFF_BHH = FOFF_BIH + 3 * WH;      // 384 f32
constexpr int FOFF_HPF = FOFF_BHH + 3 * WH;      // 16*128 f32
constexpr int FOFF_XG  = FOFF_HPF + 16 * WH;     // 16*384 f32
constexpr int FOFF_HG  = FOFF_XG + 16 * 3 * WH;  // 16*384 f32
constexpr int NF       = FOFF_HG + 16 * 3 * WH;  // = 15104 f32 (60416 B)

constexpr size_t SCAN_LDS_BYTES = (size_t)NBF * 2 + (size_t)NF * 4; // ~247 KB

__global__ void __launch_bounds__(256)
gru_scan_kernel(const bf16* __restrict__ Xb, const float* __restrict__ h0,
                const bf16* __restrict__ WihB, const bf16* __restrict__ WhhB,
                const bf16* __restrict__ WrecB,
                const float* __restrict__ b_ih, const float* __restrict__ b_hh,
                float* __restrict__ hidden) {
  extern __shared__ char smem[];
  bf16*  sBF = reinterpret_cast<bf16*>(smem);
  float* sF  = reinterpret_cast<float*>(smem + (size_t)NBF * 2);

  bf16* sWrec = sBF + OFF_WREC;
  bf16* sWhh  = sBF + OFF_WHH;
  bf16* sWih  = sBF + OFF_WIH;
  bf16* sH    = sBF + OFF_H;
  bf16* sHpB  = sBF + OFF_HPB;
  bf16* sXt   = sBF + OFF_XT;     // two buffers of 16*WNI

  float* sBih = sF + FOFF_BIH;
  float* sBhh = sF + FOFF_BHH;
  float* sHpF = sF + FOFF_HPF;
  float* sXg  = sF + FOFF_XG;
  float* sHg  = sF + FOFF_HG;

  const int lane = threadIdx.x;       // 0..31
  const int w    = threadIdx.y;       // wave id 0..7
  const int tid  = w * 32 + lane;
  const int b0   = blockIdx.x * 16;   // 16 batch rows per WG

  // Per-thread slice for async X staging: 256 threads x 8 B = 2 KB tile.
  const int xr = tid >> 4;            // row 0..15
  const int xc = (tid & 15) * 4;      // bf16 col, step 4 (8 bytes)
  const bf16* xrow = Xb + (size_t)(b0 + xr) * WT * WNI + xc;

  // --- one-time: stage weights/biases into LDS, init h, kick async X(t=0) ---
  async_copy_b64(xrow /* t = 0 */, sXt + xr * WNI + xc);

  for (int i = tid; i < WH * WH; i += 256)      sWrec[i] = WrecB[i];
  for (int i = tid; i < 3 * WH * WH; i += 256)  sWhh[i]  = WhhB[i];
  for (int i = tid; i < 3 * WH * WNI; i += 256) sWih[i]  = WihB[i];
  for (int i = tid; i < 3 * WH; i += 256) { sBih[i] = b_ih[i]; sBhh[i] = b_hh[i]; }
  for (int i = tid; i < 16 * WH; i += 256)
    sH[i] = (bf16)h0[(size_t)(b0 + (i >> 7)) * WH + (i & 127)];

  for (int t = 0; t < WT; ++t) {
    wait_async0();       // this wave's async X tile is in LDS
    __syncthreads();     // all waves' tiles + previous phase-3 writes visible

    bf16* sXcur = sXt + (t & 1) * (16 * WNI);

    // ---- phase 1: hp = h @ Wrec^T (critical chain) + xg = Xt @ Wih^T ----
    {
      v8f acc = {};
#pragma unroll
      for (int ks = 0; ks < 4; ++ks) {
        v16bf a  = ld_tile16(sH + ks * 32, WH, lane);
        v16bf bm = ld_tile16(sWrec + (w * 16) * WH + ks * 32, WH, lane);
        acc = wmma_bf16(a, bm, acc);
      }
      st_tile_f32(sHpF + w * 16, WH, lane, acc);
      st_tile_bf16(sHpB + w * 16, WH, lane, acc);
    }
#pragma unroll
    for (int j = 0; j < 3; ++j) {
      const int nt = w + 8 * j;     // 24 n-tiles over 8 waves
      v8f acc = {};
#pragma unroll
      for (int ks = 0; ks < 2; ++ks) {
        v16bf a  = ld_tile16(sXcur + ks * 32, WNI, lane);
        v16bf bm = ld_tile16(sWih + (nt * 16) * WNI + ks * 32, WNI, lane);
        acc = wmma_bf16(a, bm, acc);
      }
      st_tile_f32(sXg + nt * 16, 3 * WH, lane, acc);
    }
    __syncthreads();

    // X tile for t is consumed; overlap the t+1 copy with phases 2/3.
    if (t + 1 < WT) {
      async_copy_b64(xrow + (size_t)(t + 1) * WNI,
                     sXt + ((t + 1) & 1) * (16 * WNI) + xr * WNI + xc);
      if (t + 2 < WT)
        __builtin_prefetch(xrow + (size_t)(t + 2) * WNI, 0, 0);
    }

    // ---- phase 2: hg = hp @ Whh^T ----
#pragma unroll
    for (int j = 0; j < 3; ++j) {
      const int nt = w + 8 * j;
      v8f acc = {};
#pragma unroll
      for (int ks = 0; ks < 4; ++ks) {
        v16bf a  = ld_tile16(sHpB + ks * 32, WH, lane);
        v16bf bm = ld_tile16(sWhh + (nt * 16) * WH + ks * 32, WH, lane);
        acc = wmma_bf16(a, bm, acc);
      }
      st_tile_f32(sHg + nt * 16, 3 * WH, lane, acc);
    }
    __syncthreads();

    // ---- phase 3: gates + state update ----
    for (int i = tid; i < 16 * WH; i += 256) {
      const int r = i >> 7, c = i & 127;
      const float xg_r = sXg[r * 384 + c]       + sBih[c];
      const float xg_z = sXg[r * 384 + 128 + c] + sBih[128 + c];
      const float xg_n = sXg[r * 384 + 256 + c] + sBih[256 + c];
      const float hg_r = sHg[r * 384 + c]       + sBhh[c];
      const float hg_z = sHg[r * 384 + 128 + c] + sBhh[128 + c];
      const float hg_n = sHg[r * 384 + 256 + c] + sBhh[256 + c];
      const float rg = sigmoidf_fast(xg_r + hg_r);
      const float zg = sigmoidf_fast(xg_z + hg_z);
      const float ng = tanhf(xg_n + rg * hg_n);
      const float hp = sHpF[i];
      const float hnew = (1.0f - zg) * ng + zg * hp;   // leak uses W(h)
      hidden[((size_t)(b0 + r) * WT + t) * WH + c] = hnew;
      sH[i] = (bf16)hnew;
    }
    __syncthreads();
  }
}

// ---------------- Kernel 3: output projection ----------------
// out[M, NO] = hidden[M, H] @ W_out^T + b_out ; M = B*T.

__global__ void __launch_bounds__(256)
outproj_kernel(const float* __restrict__ hidden, const bf16* __restrict__ WoutB,
               const float* __restrict__ b_out, float* __restrict__ out) {
  __shared__ bf16  sW[WNO * WH];   // 16 KB
  __shared__ float sB[WNO];

  const int lane = threadIdx.x;
  const int w    = threadIdx.y;
  const int tid  = w * 32 + lane;

  for (int i = tid; i < WNO * WH; i += 256) sW[i] = WoutB[i];
  for (int i = tid; i < WNO; i += 256)      sB[i] = b_out[i];
  __syncthreads();

  const size_t m0 = ((size_t)blockIdx.x * 8 + w) * 16;

  v8f acc0 = {}, acc1 = {}, acc2 = {}, acc3 = {};
#pragma unroll
  for (int ks = 0; ks < 4; ++ks) {
    const float* ap =
        hidden + (m0 + (lane & 15)) * WH + ks * 32 + ((lane >> 4) << 4);
    v16bf a;
#pragma unroll
    for (int i = 0; i < 16; ++i) a[i] = (bf16)ap[i];
    acc0 = wmma_bf16(a, ld_tile16(sW + 0 * 16 * WH + ks * 32, WH, lane), acc0);
    acc1 = wmma_bf16(a, ld_tile16(sW + 1 * 16 * WH + ks * 32, WH, lane), acc1);
    acc2 = wmma_bf16(a, ld_tile16(sW + 2 * 16 * WH + ks * 32, WH, lane), acc2);
    acc3 = wmma_bf16(a, ld_tile16(sW + 3 * 16 * WH + ks * 32, WH, lane), acc3);
  }

  const int col = lane & 15;
  const int r0  = (lane >> 4) << 3;
  v8f accs[4] = {acc0, acc1, acc2, acc3};
#pragma unroll
  for (int nt = 0; nt < 4; ++nt) {
    const int n = nt * 16 + col;
    const float bias = sB[n];
#pragma unroll
    for (int i = 0; i < 8; ++i)
      out[(m0 + r0 + i) * WNO + n] = accs[nt][i] + bias;
  }
}

// ---------------- host launcher ----------------

extern "C" void kernel_launch(void* const* d_in, const int* in_sizes, int n_in,
                              void* d_out, int out_size, void* d_ws, size_t ws_size,
                              hipStream_t stream) {
  const float* X     = (const float*)d_in[0];
  const float* h0    = (const float*)d_in[1];
  const float* W_ih  = (const float*)d_in[2];
  const float* W_hh  = (const float*)d_in[3];
  const float* b_ih  = (const float*)d_in[4];
  const float* b_hh  = (const float*)d_in[5];
  const float* W_rec = (const float*)d_in[6];
  const float* W_out = (const float*)d_in[7];
  const float* b_out = (const float*)d_in[8];

  float* out    = (float*)d_out;                         // [B,T,NO]
  float* hidden = (float*)d_out + (size_t)WB * WT * WNO; // [B,T,H]

  // workspace: X in bf16 (33.5 MB) then bf16 weights
  bf16* XB    = (bf16*)d_ws;
  bf16* WihB  = XB + (size_t)WB * WT * WNI;
  bf16* WhhB  = WihB + 3 * WH * WNI;
  bf16* WrecB = WhhB + 3 * WH * WH;
  bf16* WoutB = WrecB + WH * WH;

  // 1) bf16 staging (X vectorized, weights scalar)
  const int xn4 = (WB * WT * WNI) / 4;
  x_to_bf16_kernel<<<(xn4 + 255) / 256, 256, 0, stream>>>(X, XB, xn4);
  f32_to_bf16_kernel<<<(3 * WH * WNI + 255) / 256, 256, 0, stream>>>(W_ih, WihB, 3 * WH * WNI);
  f32_to_bf16_kernel<<<(3 * WH * WH  + 255) / 256, 256, 0, stream>>>(W_hh, WhhB, 3 * WH * WH);
  f32_to_bf16_kernel<<<(WH * WH      + 255) / 256, 256, 0, stream>>>(W_rec, WrecB, WH * WH);
  f32_to_bf16_kernel<<<(WNO * WH     + 255) / 256, 256, 0, stream>>>(W_out, WoutB, WNO * WH);

  // 2) fused recurrent scan (dynamic LDS ~247 KB of the 320 KB/WGP)
  (void)hipFuncSetAttribute((const void*)gru_scan_kernel,
                            hipFuncAttributeMaxDynamicSharedMemorySize,
                            (int)SCAN_LDS_BYTES);
  gru_scan_kernel<<<dim3(WB / 16), dim3(32, 8), SCAN_LDS_BYTES, stream>>>(
      XB, h0, WihB, WhhB, WrecB, b_ih, b_hh, hidden);

  // 3) output projection GEMM
  outproj_kernel<<<dim3((WB * WT / 16) / 8), dim3(32, 8), 0, stream>>>(
      hidden, WoutB, b_out, out);
}
